// Transformer_66554813219292
// MI455X (gfx1250) — matrix operand order repaired and checked by reading, not compiled, not run on hardware
//
#include <hip/hip_runtime.h>

// ---------------- problem constants ----------------
constexpr int Bb = 16, Ss = 100, Tt = 60, Dd = 512, Hh = 8, Vv = 32000, Rr = 20;
constexpr int DKk = Dd / Hh;      // 64
constexpr int WINw = 20;
constexpr int VR = Vv + Rr;       // 32020
static_assert(Dd == 512, "shift tricks assume D=512");

#ifndef __has_builtin
#define __has_builtin(x) 0
#endif

// ---------------- CDNA5 async global->LDS copy (ASYNCcnt path) ----------------
#if __has_builtin(__builtin_amdgcn_global_load_async_to_lds_b128)
#define HAVE_ASYNC 1
#else
#define HAVE_ASYNC 0
#endif

// Builtin signature (from compiler diagnostic): first param is a pointer to a
// GCC-vector int4 in the global address space; second is the LDS destination.
typedef int v4i_vs __attribute__((vector_size(16)));

__device__ inline void async_copy16(const float* g, float* l) {
#if HAVE_ASYNC
  __builtin_amdgcn_global_load_async_to_lds_b128(
      (__attribute__((address_space(1))) v4i_vs*)(uintptr_t)g,
      (__attribute__((address_space(3))) v4i_vs*)(unsigned)(uintptr_t)l,
      0, 0);
#else
  *(float4*)l = *(const float4*)g;
#endif
}

#if HAVE_ASYNC
#if __has_builtin(__builtin_amdgcn_s_wait_asynccnt)
#define WAIT_ASYNC(n) __builtin_amdgcn_s_wait_asynccnt(n)
#else
#define WAIT_ASYNC(n) asm volatile("s_wait_asynccnt %0" ::"i"(n) : "memory")
#endif
#else
#define WAIT_ASYNC(n) ((void)0)
#endif

// ---------------- small helpers ----------------
__device__ inline float wave_reduce_sum(float x) {
  #pragma unroll
  for (int o = 16; o; o >>= 1) x += __shfl_xor(x, o, 32);
  return x;
}
__device__ inline float wave_reduce_max(float x) {
  #pragma unroll
  for (int o = 16; o; o >>= 1) x = fmaxf(x, __shfl_xor(x, o, 32));
  return x;
}
__device__ inline float sigmoidf(float x) { return 1.0f / (1.0f + expf(-x)); }

// ---------------- K1: masked softmax over S per (t,b) ----------------
__global__ void attn_softmax_kernel(const float* __restrict__ logits, // [B,T,S]
                                    const unsigned char* __restrict__ mask, // [B,S]
                                    float* __restrict__ attn) {      // [T,B,S]
  int row = blockIdx.x * (blockDim.x >> 5) + (threadIdx.x >> 5);
  if (row >= Tt * Bb) return;
  int t = row / Bb, b = row % Bb;
  int lane = threadIdx.x & 31;
  float vals[4];
  float mx = -1e30f;
  #pragma unroll
  for (int i = 0; i < 4; ++i) {
    int s = lane + 32 * i;
    float l = -1e30f;
    if (s < Ss && mask[b * Ss + s]) l = logits[((size_t)b * Tt + t) * Ss + s];
    vals[i] = l;
    mx = fmaxf(mx, l);
  }
  mx = wave_reduce_max(mx);
  float sum = 0.0f;
  #pragma unroll
  for (int i = 0; i < 4; ++i) {
    vals[i] = (vals[i] > -1e29f) ? expf(vals[i] - mx) : 0.0f;
    sum += vals[i];
  }
  sum = wave_reduce_sum(sum);
  float inv = 1.0f / sum;
  #pragma unroll
  for (int i = 0; i < 4; ++i) {
    int s = lane + 32 * i;
    if (s < Ss) attn[(size_t)row * Ss + s] = vals[i] * inv;
  }
}

// ---------------- K2a: decay-windowed attn sum ----------------
__global__ void window_attn_kernel(const float* __restrict__ attn, // [T,B,S]
                                   float* __restrict__ attnw) {    // [T,B,S]
  int t = blockIdx.x;
  int kmin = max(0, (WINw - 1) - t);
  float denom = 0.0f;
  for (int j = kmin; j < WINw; ++j) denom += expf((j + 1) * 0.05f);
  float inv = 1.0f / denom;
  for (int idx = threadIdx.x; idx < Bb * Ss; idx += blockDim.x) {
    float acc = 0.0f;
    for (int k = kmin; k < WINw; ++k) {
      int tau = t - (WINw - 1) + k;
      acc += expf((k + 1) * 0.05f) * inv * attn[(size_t)tau * Bb * Ss + idx];
    }
    attnw[(size_t)t * Bb * Ss + idx] = acc;
  }
}

// ---------------- K2b: decay-windowed query sum ----------------
__global__ void window_query_kernel(const float* __restrict__ q,  // [B,T,D]
                                    float* __restrict__ qw) {     // [T,B,D]
  int t = blockIdx.x;
  int kmin = max(0, (WINw - 1) - t);
  float denom = 0.0f;
  for (int j = kmin; j < WINw; ++j) denom += expf((j + 1) * 0.05f);
  float inv = 1.0f / denom;
  for (int idx = threadIdx.x; idx < Bb * Dd; idx += blockDim.x) {
    int b = idx >> 9, d = idx & 511;
    float acc = 0.0f;
    for (int k = kmin; k < WINw; ++k) {
      int tau = t - (WINw - 1) + k;
      acc += expf((k + 1) * 0.05f) * inv * q[((size_t)b * Tt + tau) * Dd + d];
    }
    qw[(size_t)t * Bb * Dd + idx] = acc;
  }
}

// ---------------- K3: per-(t,b) scalar gates rem/addp/pgen ----------------
__global__ void scalars_kernel(const float* __restrict__ q,   // [B,T,D]
                               const float* __restrict__ Wr, const float* __restrict__ br,
                               const float* __restrict__ Wa, const float* __restrict__ ba,
                               const float* __restrict__ Wptr, const float* __restrict__ bptr,
                               float* __restrict__ remv, float* __restrict__ addpv,
                               float* __restrict__ pgenv) {
  int row = blockIdx.x * (blockDim.x >> 5) + (threadIdx.x >> 5);
  if (row >= Tt * Bb) return;
  int t = row / Bb, b = row % Bb;
  int lane = threadIdx.x & 31;
  const float* qp = q + ((size_t)b * Tt + t) * Dd;
  float pr = 0, pa = 0, pp = 0;
  for (int d = lane; d < Dd; d += 32) {
    float x = qp[d];
    pr += x * Wr[d]; pa += x * Wa[d]; pp += x * Wptr[d];
  }
  pr = wave_reduce_sum(pr); pa = wave_reduce_sum(pa); pp = wave_reduce_sum(pp);
  if (lane == 0) {
    remv[row]  = sigmoidf(pr + br[0]);
    addpv[row] = sigmoidf(pa + ba[0]);
    pgenv[row] = sigmoidf(pp + bptr[0]);
  }
}

// ---------------- K4: last_ctx[t,b,d] = sum_s attnw * v ----------------
__global__ void last_ctx_kernel(const float* __restrict__ attnw, // [T,B,S]
                                const float* __restrict__ v,     // [B,S,D]
                                float* __restrict__ lastctx) {   // [T,B,D]
  int t = blockIdx.x;
  for (int idx = threadIdx.x; idx < Bb * Dd; idx += blockDim.x) {
    int b = idx >> 9, d = idx & 511;
    const float* aw = attnw + ((size_t)t * Bb + b) * Ss;
    const float* vp = v + (size_t)b * Ss * Dd + d;
    float acc = 0.0f;
    for (int s = 0; s < Ss; ++s) acc += aw[s] * vp[(size_t)s * Dd];
    lastctx[(size_t)t * Bb * Dd + idx] = acc;
  }
}

// ---------------- K5: vn[b,s] = v . Wn[0:D] ----------------
__global__ void vn_kernel(const float* __restrict__ v, const float* __restrict__ Wn,
                          float* __restrict__ vn) {
  int row = blockIdx.x * (blockDim.x >> 5) + (threadIdx.x >> 5);
  if (row >= Bb * Ss) return;
  int lane = threadIdx.x & 31;
  const float* vp = v + (size_t)row * Dd;
  float p = 0;
  for (int d = lane; d < Dd; d += 32) p += vp[d] * Wn[d];
  p = wave_reduce_sum(p);
  if (lane == 0) vn[row] = p;
}

// ---------------- K6: ln[t,b] = last_ctx . Wn[D:2D] ----------------
__global__ void ln_kernel(const float* __restrict__ lastctx, const float* __restrict__ Wn,
                          float* __restrict__ lnv) {
  int row = blockIdx.x * (blockDim.x >> 5) + (threadIdx.x >> 5);
  if (row >= Tt * Bb) return;
  int lane = threadIdx.x & 31;
  const float* lp = lastctx + (size_t)row * Dd;
  float p = 0;
  for (int d = lane; d < Dd; d += 32) p += lp[d] * Wn[Dd + d];
  p = wave_reduce_sum(p);
  if (lane == 0) lnv[row] = p;
}

// ---------------- K7: gain[t,b,s] ----------------
__global__ void gain_kernel(const float* __restrict__ vn, const float* __restrict__ lnv,
                            const float* __restrict__ addpv, const float* __restrict__ bn,
                            float* __restrict__ gain) {
  int idx = blockIdx.x * blockDim.x + threadIdx.x;
  if (idx >= Tt * Bb * Ss) return;
  int t = idx / (Bb * Ss);
  int r = idx % (Bb * Ss);
  int b = r / Ss;
  float nxt = sigmoidf(vn[r] + lnv[t * Bb + b] + bn[0]);
  gain[idx] = addpv[t * Bb + b] * nxt;
}

// ---------------- K8: (1 - add_state) prefix product ----------------
__global__ void a1m_kernel(const float* __restrict__ gain, float* __restrict__ a1m) {
  int idx = blockIdx.x * blockDim.x + threadIdx.x;
  if (idx >= Bb * Ss) return;
  int s = idx % Ss;
  float A = (s < 50) ? ((s + 1) * 0.02f) : 1.0f;  // 1 - add0[s]
  for (int t = 0; t < Tt; ++t) {
    a1m[(size_t)t * Bb * Ss + idx] = A;
    A *= (1.0f - gain[(size_t)t * Bb * Ss + idx]);
  }
}

// ---------------- WMMA types ----------------
typedef __attribute__((ext_vector_type(16))) __bf16 v16bf;
typedef __attribute__((ext_vector_type(8)))  float  v8f;

// ---------------- generic single-wave-tile bf16 WMMA GEMM (small GEMMs) ----
// C[m,n] = sum_k A[m,k]*Bw[n*sN+k*sK] + bias[n]
__global__ __launch_bounds__(128)
void gemm_bf16_kernel(const float* __restrict__ A, int lda,
                      const float* __restrict__ Bw, long sN, long sK,
                      const float* __restrict__ bias,
                      float* __restrict__ C, long ldc, int K) {
  const int lane = threadIdx.x & 31;
  const int wave = threadIdx.x >> 5;
  const int n0 = (blockIdx.x * (blockDim.x >> 5) + wave) * 16;
  const int m0 = blockIdx.y * 16;
  const int half = lane >> 4;
  const int l15 = lane & 15;
  const float* arow = A + (size_t)(m0 + l15) * lda;
  const float* bcol = Bw + (size_t)(n0 + l15) * sN;
  v8f c;
  float bv = bias ? bias[n0 + l15] : 0.0f;
  #pragma unroll
  for (int j = 0; j < 8; ++j) c[j] = bv;
  for (int k0 = 0; k0 < K; k0 += 32) {
    __builtin_prefetch(arow + k0 + 32, 0, 1);
    v16bf a, b;
    #pragma unroll
    for (int i = 0; i < 8; ++i) {
      a[i]     = (__bf16)arow[k0 + 8 * half + i];
      a[8 + i] = (__bf16)arow[k0 + 16 + 8 * half + i];
    }
    #pragma unroll
    for (int i = 0; i < 8; ++i) {
      b[i]     = (__bf16)bcol[(size_t)(k0 + 8 * half + i) * sK];
      b[8 + i] = (__bf16)bcol[(size_t)(k0 + 16 + 8 * half + i) * sK];
    }
    c = __builtin_amdgcn_wmma_f32_16x16x32_bf16(false, a, false, b, (short)0, c,
                                                false, false);
  }
  #pragma unroll
  for (int j = 0; j < 8; ++j)
    C[(size_t)(m0 + 8 * half + j) * ldc + (n0 + l15)] = c[j];
}

// ---------------- vocab GEMM: async-LDS staged, double-buffered ------------
// out[m, n] = sum_k A[m,k] * Bw[n,k] + bias[n];  A:[960,512], Bw:[32000,512]
constexpr int BM = 32, BN = 256, BK = 32;
constexpr int LSTR = BK + 4;              // padded row stride (floats), 16B aligned
constexpr int NSTEP = Dd / BK;            // 16 k-steps
constexpr size_t VOCAB_LDS =
    (size_t)(2 * BM * LSTR + 2 * BN * LSTR) * sizeof(float);  // ~83 KB

__global__ __launch_bounds__(256)
void vocab_gemm_kernel(const float* __restrict__ A,
                       const float* __restrict__ Bw,
                       const float* __restrict__ bias,
                       float* __restrict__ C, long ldc) {
  extern __shared__ float sm[];
  float* smA = sm;                        // 2 * BM*LSTR
  float* smB = sm + 2 * BM * LSTR;        // 2 * BN*LSTR
  const int tid = threadIdx.x;
  const int lane = tid & 31, wave = tid >> 5;
  const int wm = wave >> 2, wn = wave & 3;   // 2 (M) x 4 (N) waves
  const int m0 = blockIdx.y * BM;
  const int n0 = blockIdx.x * BN;
  const int half = lane >> 4, l15 = lane & 15;
  const int arow = tid >> 3, achunk = tid & 7;  // A: 32 rows x 8 chunks, 1/thread

  const float* Ag = A + (size_t)(m0 + arow) * Dd + achunk * 4;
  const float* Bg = Bw + (size_t)(n0 + tid) * Dd;  // B: one row per thread

  auto stage = [&](int buf, int k0) {
    async_copy16(Ag + k0, smA + buf * BM * LSTR + arow * LSTR + achunk * 4);
    float* lb = smB + buf * BN * LSTR + tid * LSTR;
    #pragma unroll
    for (int cc = 0; cc < 8; ++cc)
      async_copy16(Bg + k0 + cc * 4, lb + cc * 4);
  };

  v8f c[4];
  #pragma unroll
  for (int f = 0; f < 4; ++f) {
    float bv = bias[n0 + wn * 64 + f * 16 + l15];
    #pragma unroll
    for (int j = 0; j < 8; ++j) c[f][j] = bv;
  }

  stage(0, 0);  // 9 async loads per thread outstanding
  for (int ks = 0; ks < NSTEP; ++ks) {
    const int buf = ks & 1;
    if (ks + 1 < NSTEP) {
      stage(buf ^ 1, (ks + 1) * BK);  // prefetch next tile (9 more outstanding)
      WAIT_ASYNC(9);                  // current tile's 9 loads complete (in-order)
    } else {
      WAIT_ASYNC(0);
    }
    __syncthreads();

    const float* Ar = smA + buf * BM * LSTR + (wm * 16 + l15) * LSTR;
    v16bf a;
    #pragma unroll
    for (int i = 0; i < 8; ++i) {
      a[i]     = (__bf16)Ar[8 * half + i];
      a[8 + i] = (__bf16)Ar[16 + 8 * half + i];
    }
    #pragma unroll
    for (int f = 0; f < 4; ++f) {
      const float* Br =
          smB + buf * BN * LSTR + (wn * 64 + f * 16 + l15) * LSTR;
      v16bf b;
      #pragma unroll
      for (int i = 0; i < 8; ++i) {
        b[i]     = (__bf16)Br[8 * half + i];
        b[8 + i] = (__bf16)Br[16 + 8 * half + i];
      }
      c[f] = __builtin_amdgcn_wmma_f32_16x16x32_bf16(false, a, false, b,
                                                     (short)0, c[f], false, false);
    }
    __syncthreads();  // all waves done reading buf before it is restaged
  }
  #pragma unroll
  for (int f = 0; f < 4; ++f)
    #pragma unroll
    for (int j = 0; j < 8; ++j)
      C[(size_t)(m0 + wm * 16 + 8 * half + j) * ldc +
        (n0 + wn * 64 + f * 16 + l15)] = c[f][j];
}

// ---------------- K9: the sequential recurrence, one workgroup per batch ----
__global__ __launch_bounds__(512)
void seq_step_kernel(const float* __restrict__ v,      // [B,S,D]
                     const float* __restrict__ attn,   // [T,B,S]
                     const float* __restrict__ attnw,  // [T,B,S]
                     const float* __restrict__ u,      // [T,B,D]
                     const float* __restrict__ gain,   // [T,B,S]
                     const float* __restrict__ a1m,    // [T,B,S]
                     const float* __restrict__ remv,   // [T,B]
                     const float* __restrict__ pgenv,  // [T,B]
                     const float* __restrict__ kheads, // [B,R,D]
                     const float* __restrict__ Wqr,    // [D,D]
                     const float* __restrict__ bqr,    // [D]
                     float* __restrict__ out) {        // [B,T,V+R]
  extern __shared__ float sm[];
  float* mem    = sm;                    // S*D
  float* sel    = mem + Ss * Dd;         // D
  float* qh     = sel + Dd;              // D
  float* uS     = qh + Dd;               // D
  float* attnS  = uS + Dd;               // S
  float* attnwS = attnS + Ss;            // S
  float* gainS  = attnwS + Ss;           // S
  float* a1mS   = gainS + Ss;            // S
  float* fac    = a1mS + Ss;             // S
  float* scores = fac + Ss;              // H*R

  const int b = blockIdx.x;
  const int tid = threadIdx.x;
  const int nt = blockDim.x;
  const float* vb = v + (size_t)b * Ss * Dd;

  for (int i = tid; i < Ss * Dd; i += nt) {
    int s = i >> 9;
    float add0 = (s < 50) ? (1.0f - (s + 1) * 0.02f) : 0.0f;
    mem[i] = vb[i] * add0;
  }
  __syncthreads();

  for (int t = 0; t < Tt; ++t) {
    const size_t rbs = (size_t)t * Bb + b;
    for (int i = tid; i < Ss; i += nt) {
      attnS[i]  = attn [rbs * Ss + i];
      attnwS[i] = attnw[rbs * Ss + i];
      gainS[i]  = gain [rbs * Ss + i];
      a1mS[i]   = a1m  [rbs * Ss + i];
    }
    for (int i = tid; i < Dd; i += nt) uS[i] = u[rbs * Dd + i];
    __syncthreads();

    for (int d = tid; d < Dd; d += nt) {
      float acc = 0.0f;
      #pragma unroll 4
      for (int s = 0; s < Ss; ++s) acc += attnS[s] * mem[s * Dd + d];
      sel[d] = acc;
    }
    __syncthreads();

    for (int d = tid; d < Dd; d += nt) {
      float acc = bqr[d];
      for (int e = 0; e < Dd; ++e) acc += sel[e] * Wqr[(size_t)e * Dd + d];
      qh[d] = acc;
    }
    __syncthreads();

    if (tid < Hh * Rr) {
      int h = tid / Rr, r = tid % Rr;
      const float* kp = kheads + ((size_t)b * Rr + r) * Dd + h * DKk;
      const float* qp = qh + h * DKk;
      float acc = 0.0f;
      for (int k = 0; k < DKk; ++k) acc += qp[k] * kp[k];
      scores[tid] = acc * 0.125f;  // / sqrt(64)
    }
    __syncthreads();
    if (tid < Hh) {
      int h = tid;
      float mx = -1e30f;
      for (int r = 0; r < Rr; ++r) mx = fmaxf(mx, scores[h * Rr + r]);
      float dn = 0.0f;
      for (int r = 0; r < Rr; ++r) dn += expf(scores[h * Rr + r] - mx);
      float inv = 1.0f / dn;
      for (int r = 0; r < Rr; ++r)
        scores[h * Rr + r] = expf(scores[h * Rr + r] - mx) * inv;
    }
    __syncthreads();
    if (tid < Rr) {
      float acc = 0.0f;
      for (int h = 0; h < Hh; ++h) acc += scores[h * Rr + tid];
      float ra = acc * (1.0f / Hh);
      out[((size_t)b * Tt + t) * VR + Vv + tid] = ra * pgenv[rbs];
    }

    {
      float rem_b = remv[rbs];
      int wave = tid >> 5, lane = tid & 31, nw = nt >> 5;
      for (int s = wave; s < Ss; s += nw) {
        float p = 0.0f;
        for (int d = lane; d < Dd; d += 32) p += mem[s * Dd + d] * uS[d];
        p = wave_reduce_sum(p);
        if (lane == 0) fac[s] = 1.0f - rem_b * attnwS[s] * sigmoidf(p);
      }
    }
    __syncthreads();

    for (int i = tid; i < Ss * Dd; i += nt) {
      int s = i >> 9;
      mem[i] = mem[i] * fac[s] + vb[i] * (a1mS[s] * gainS[s]);
    }
    __syncthreads();
  }
}

// ---------------- host launcher ----------------
extern "C" void kernel_launch(void* const* d_in, const int* in_sizes, int n_in,
                              void* d_out, int out_size, void* d_ws, size_t ws_size,
                              hipStream_t stream) {
  const float* v_outputs   = (const float*)d_in[0];
  const float* f_outputs   = (const float*)d_in[1];
  const float* query_seq   = (const float*)d_in[2];
  const float* attn_logits = (const float*)d_in[3];
  const float* Wr     = (const float*)d_in[4];
  const float* br     = (const float*)d_in[5];
  const float* Wa     = (const float*)d_in[6];
  const float* ba     = (const float*)d_in[7];
  const float* Wq_lin = (const float*)d_in[8];
  const float* Wn     = (const float*)d_in[9];
  const float* bn     = (const float*)d_in[10];
  const float* Wlogit = (const float*)d_in[11];
  const float* blogit = (const float*)d_in[12];
  const float* Wptr   = (const float*)d_in[13];
  const float* bptr   = (const float*)d_in[14];
  const float* Wqr    = (const float*)d_in[15];
  const float* bqr    = (const float*)d_in[16];
  const float* Wkr    = (const float*)d_in[17];
  const float* bkr    = (const float*)d_in[18];
  const unsigned char* src_mask = (const unsigned char*)d_in[19];
  float* out = (float*)d_out;

  float* ws = (float*)d_ws;
  size_t o = 0;
  float* attn    = ws + o; o += (size_t)Tt * Bb * Ss;
  float* attnw   = ws + o; o += (size_t)Tt * Bb * Ss;
  float* qw      = ws + o; o += (size_t)Tt * Bb * Dd;
  float* u       = ws + o; o += (size_t)Tt * Bb * Dd;
  float* lastctx = ws + o; o += (size_t)Tt * Bb * Dd;
  float* gain    = ws + o; o += (size_t)Tt * Bb * Ss;
  float* a1m     = ws + o; o += (size_t)Tt * Bb * Ss;
  float* remv    = ws + o; o += (size_t)Tt * Bb;
  float* addpv   = ws + o; o += (size_t)Tt * Bb;
  float* pgenv   = ws + o; o += (size_t)Tt * Bb;
  float* vn      = ws + o; o += (size_t)Bb * Ss;
  float* lnv     = ws + o; o += (size_t)Tt * Bb;
  float* kheads  = ws + o; o += (size_t)Bb * Rr * Dd;

  attn_softmax_kernel<<<dim3((Tt * Bb + 3) / 4), dim3(128), 0, stream>>>(
      attn_logits, src_mask, attn);
  window_attn_kernel<<<dim3(Tt), dim3(256), 0, stream>>>(attn, attnw);
  window_query_kernel<<<dim3(Tt), dim3(256), 0, stream>>>(query_seq, qw);
  scalars_kernel<<<dim3((Tt * Bb + 3) / 4), dim3(128), 0, stream>>>(
      query_seq, Wr, br, Wa, ba, Wptr, bptr, remv, addpv, pgenv);
  last_ctx_kernel<<<dim3(Tt), dim3(256), 0, stream>>>(attnw, v_outputs, lastctx);
  vn_kernel<<<dim3((Bb * Ss + 3) / 4), dim3(128), 0, stream>>>(v_outputs, Wn, vn);
  ln_kernel<<<dim3((Tt * Bb + 3) / 4), dim3(128), 0, stream>>>(lastctx, Wn, lnv);
  gain_kernel<<<dim3((Tt * Bb * Ss + 255) / 256), dim3(256), 0, stream>>>(
      vn, lnv, addpv, bn, gain);
  a1m_kernel<<<dim3((Bb * Ss + 255) / 256), dim3(256), 0, stream>>>(gain, a1m);

  // small WMMA GEMMs (direct-load single-wave tiles)
  gemm_bf16_kernel<<<dim3(Dd / 16 / 4, (Bb * Rr) / 16), dim3(128), 0, stream>>>(
      f_outputs, Dd, Wkr, 1L, (long)Dd, bkr, kheads, (long)Dd, Dd);
  gemm_bf16_kernel<<<dim3(Dd / 16 / 4, (Tt * Bb) / 16), dim3(128), 0, stream>>>(
      qw, Dd, Wq_lin, (long)Dd, 1L, nullptr, u, (long)Dd, Dd);

  // vocab GEMM: async-LDS double-buffered, 4 WMMA per wave per k-step
  vocab_gemm_kernel<<<dim3(Vv / BN, (Bb * Tt) / BM), dim3(256), VOCAB_LDS,
                      stream>>>(query_seq, Wlogit, blogit, out, (long)VR);

  // sequential recurrence: one WGP per batch, mem resident in ~214KB LDS
  size_t smem = (size_t)(Ss * Dd + 3 * Dd + 5 * Ss + Hh * Rr) * sizeof(float);
  seq_step_kernel<<<dim3(Bb), dim3(512), smem, stream>>>(
      v_outputs, attn, attnw, u, gain, a1m, remv, pgenv, kheads, Wqr, bqr, out);
}
